// StereoBMLoss_28278064677095
// MI455X (gfx1250) — matrix operand order repaired and checked by reading, not compiled, not run on hardware
//
#include <hip/hip_runtime.h>
#include <hip/hip_bf16.h>

// ---------------------------------------------------------------------------
// StereoBM loss on gfx1250: 21x21 box filter expressed as banded-ones matrix
// products executed with v_wmma_f32_16x16x32_f16. SAD stage uses packed-f16
// VALU with the left-image strip held in VGPRs across the disparity loop.
// ---------------------------------------------------------------------------

typedef _Float16 h4  __attribute__((ext_vector_type(4)));
typedef _Float16 h8  __attribute__((ext_vector_type(8)));
typedef _Float16 h16 __attribute__((ext_vector_type(16)));
typedef float    f8  __attribute__((ext_vector_type(8)));

#define IMGS   2
#define BATCH  2
#define HH     320
#define WW     512
#define WPAD   576                 // right image padded with 64 zero cols on the left
#define NDISP  64
#define DGROUPS 2
#define DPG    (NDISP / DGROUPS)   // disparities per workgroup
#define RADIUS 10                  // (21-1)/2
#define STRIP_ROWS 36              // 16 output rows + 2*RADIUS context
#define NPLANE (IMGS * BATCH)      // 4 independent image planes
#define PLANE_PIX (HH * WW)
#define ROW_TILES (HH / 16)        // 20

// LDS geometry (bank-conflict padded, everything 16B aligned)
#define DT_STRIDE 72                        // halfs per diff column: 64 K-rows + 8 pad
#define DT_BYTES  (WW * DT_STRIDE * 2)      // 73728
#define VB_STRIDE 560                       // 16 zero pad + 512 data + 32 zero pad
#define VB_BYTES  (16 * VB_STRIDE * 2)      // 17920
#define SMEM_BYTES (DT_BYTES + VB_BYTES)    // 91648 (<< 320KB per WGP)

#define NPART 128

// ---------------------------------------------------------------------------
// Stage 1: channel-wise L2 normalization, f32 -> f16, channel-interleaved x4.
// ---------------------------------------------------------------------------
__global__ __launch_bounds__(256)
void normalize_kernel(const float* __restrict__ x, const float* __restrict__ gt,
                      h4* __restrict__ Lp, h4* __restrict__ Rp) {
  int gid = blockIdx.x * blockDim.x + threadIdx.x;
  if (gid >= IMGS * BATCH * HH * WW) return;
  int w = gid & (WW - 1);
  int r = gid >> 9;
  int h = r % HH; r /= HH;
  int b = r & 1;
  int img = r >> 1;
  const float* src = img ? gt : x;
  const size_t hw = (size_t)HH * WW;
  size_t base = ((size_t)b * 6) * hw + (size_t)h * WW + w;
  float l0 = src[base],          l1 = src[base + hw],     l2 = src[base + 2 * hw];
  float r0 = src[base + 3 * hw], r1 = src[base + 4 * hw], r2 = src[base + 5 * hw];
  float nl = fmaxf(sqrtf(l0 * l0 + l1 * l1 + l2 * l2), 1e-12f);
  float nr = fmaxf(sqrtf(r0 * r0 + r1 * r1 + r2 * r2), 1e-12f);
  h4 lv = { (_Float16)(l0 / nl), (_Float16)(l1 / nl), (_Float16)(l2 / nl), (_Float16)0 };
  h4 rv = { (_Float16)(r0 / nr), (_Float16)(r1 / nr), (_Float16)(r2 / nr), (_Float16)0 };
  int p = img * BATCH + b;
  Lp[(size_t)p * hw + (size_t)h * WW + w] = lv;
  Rp[((size_t)p * HH + h) * WPAD + (w + 64)] = rv;
  if (w < 64) {  // zero pad so R[w-d] is branch-free and reproduces where(valid,·,0)
    h4 z = { (_Float16)0, (_Float16)0, (_Float16)0, (_Float16)0 };
    Rp[((size_t)p * HH + h) * WPAD + w] = z;
  }
}

// ---------------------------------------------------------------------------
// Stage 2: fused SAD + separable 21x21 box filter (as WMMA banded matmuls)
// + running argmin over disparities. One workgroup = one 16-row strip of one
// plane for one group of 32 disparities. 512 threads = 16 waves; wave i owns
// output column tiles 2i and 2i+1 (32 tiles cover W=512).
// ---------------------------------------------------------------------------
__global__ __launch_bounds__(512)
void bm_cost_kernel(const h4* __restrict__ Lp, const h4* __restrict__ Rp,
                    float* __restrict__ costOut, float* __restrict__ dOut) {
  extern __shared__ char smem[];
  _Float16* diffT = (_Float16*)smem;               // [WW][DT_STRIDE] transposed diff
  _Float16* vbox  = (_Float16*)(smem + DT_BYTES);  // [16][VB_STRIDE] vertical box sums

  const int tid = threadIdx.x;
  const int lane = tid & 31;
  const int wid  = tid >> 5;    // wave id 0..15
  const int lh   = lane >> 4;   // lane half (0/1)
  const int ln   = lane & 15;

  int bid = blockIdx.x;
  const int g = bid / (NPLANE * ROW_TILES);
  int rem = bid % (NPLANE * ROW_TILES);
  const int p  = rem / ROW_TILES;        // plane = img*2 + b
  const int rt = rem % ROW_TILES;
  const int h0 = rt * 16;
  const int d0 = g * DPG;

  const h4* Lpl = Lp + (size_t)p * PLANE_PIX;
  const h4* Rpl = Rp + (size_t)p * HH * WPAD;

  // Zero all LDS once: diff rows 36..63, rows outside the image, and vbox pads
  // must stay 0 (they feed WMMA; leftover poison must never reach a multiply).
  {
    unsigned* s32 = (unsigned*)smem;
    for (int i = tid; i < SMEM_BYTES / 4; i += 512) s32[i] = 0u;
  }

  // Constant banded-ones matrices, built in contiguous k order (K = 16*lh + i) for A and B.
  // Vertical A = M^T (16 x 48, padded to 2 K32 chunks): A[hh][K]=1 iff hh<=K<=hh+20.
  // Horizontal B taps: M_left[k][n]=(k-n>=6), M_center=(|n-k|<=10), M_right=(n-k>=6).
  h16 aV0, aV1, bH1, bH2;
#pragma unroll
  for (int i = 0; i < 16; ++i) {
    int kbase = lh * 16 + i;  // A half i -> K index
    int K0 = kbase, K1 = 32 + kbase;
    aV0[i] = (_Float16)((K0 >= ln && K0 <= ln + 20) ? 1.0f : 0.0f);
    aV1[i] = (_Float16)((K1 >= ln && K1 <= ln + 20) ? 1.0f : 0.0f);
    // B half i -> K = i + 16*lh, N = ln
    if (lh == 0) {
      bH1[i] = (_Float16)(((i - ln) >= 6) ? 1.0f : 0.0f);   // K 0..15  : M_left
      bH2[i] = (_Float16)(((ln - i) >= 6) ? 1.0f : 0.0f);   // K 0..15  : M_right
    } else {
      int dd = ln - i;
      bH1[i] = (_Float16)((dd <= 10 && dd >= -10) ? 1.0f : 0.0f); // K16..31: M_center
      bH2[i] = (_Float16)0;                                       // K16..31: zero
    }
  }

  // Left-image column strip is disparity-invariant: keep it in VGPRs.
  const int w = tid;  // 512 threads == WW columns
  h4 Lreg[STRIP_ROWS];
#pragma unroll
  for (int r = 0; r < STRIP_ROWS; ++r) {
    const int h = h0 - RADIUS + r;     // block-uniform bounds check
    if (0 <= h && h < HH) Lreg[r] = Lpl[h * WW + w];
  }

  f8 bestc0, bestc1, bestd0, bestd1;  // argmin state lives in VGPRs
#pragma unroll
  for (int v = 0; v < 8; ++v) {
    bestc0[v] = 3.0e38f; bestc1[v] = 3.0e38f;
    bestd0[v] = 0.0f;    bestd1[v] = 0.0f;
  }

  __syncthreads();

  for (int dd = 0; dd < DPG; ++dd) {
    const int d = d0 + dd;

    // ---- Phase 1: packed-f16 SAD for this disparity, transposed into LDS ----
    {
      _Float16* col = diffT + w * DT_STRIDE;
      // One base pointer per disparity; rows become immediate offsets.
      const h4* rbase = Rpl + (size_t)(h0 - RADIUS) * WPAD + (w + 64 - d);
#pragma unroll
      for (int r = 0; r < STRIP_ROWS; ++r) {
        const int h = h0 - RADIUS + r;  // block-uniform; OOB rows stay 0 in LDS
        if (0 <= h && h < HH) {
          h4 t = __builtin_elementwise_abs(Lreg[r] - rbase[r * WPAD]);
          col[r] = (t.x + t.y) + t.z;   // 4th channel is 0-0, ignored
        }
      }
    }
    __syncthreads();

    // ---- Phase 2: vertical box sum = (M^T 16x48) x (diff 48xW) via WMMA ----
#pragma unroll
    for (int tt = 0; tt < 2; ++tt) {
      const int j = wid * 2 + tt;
      // B-matrix: column j*16+ln of diffT, 16 contiguous halfs per K32 chunk.
      const _Float16* bp = diffT + (j * 16 + ln) * DT_STRIDE + lh * 16;
      h8 b0lo = *(const h8*)(bp);
      h8 b0hi = *(const h8*)(bp + 8);
      h8 b1lo = *(const h8*)(bp + 32);
      h8 b1hi = *(const h8*)(bp + 40);
      h16 b0 = __builtin_shufflevector(b0lo, b0hi, 0,1,2,3,4,5,6,7,8,9,10,11,12,13,14,15);
      h16 b1 = __builtin_shufflevector(b1lo, b1hi, 0,1,2,3,4,5,6,7,8,9,10,11,12,13,14,15);
      f8 c = {0.f,0.f,0.f,0.f,0.f,0.f,0.f,0.f};
      c = __builtin_amdgcn_wmma_f32_16x16x32_f16(false, aV0, false, b0, (short)0, c, false, false);
      c = __builtin_amdgcn_wmma_f32_16x16x32_f16(false, aV1, false, b1, (short)0, c, false, false);
      // D layout: row = v + 8*lh, col = ln. Store into padded vbox row buffer.
      _Float16* vp = vbox + (lh * 8) * VB_STRIDE + 16 + j * 16 + ln;
#pragma unroll
      for (int v = 0; v < 8; ++v) vp[v * VB_STRIDE] = (_Float16)c[v];
    }
    __syncthreads();

    // ---- Phase 3: horizontal box = V x M (3 taps, 2 WMMAs), update argmin ----
    f8 hc[2];
#pragma unroll
    for (int tt = 0; tt < 2; ++tt) {
      const int j = wid * 2 + tt;
      // A-matrix: row ln of vbox; K=0..31 spans tiles j-1,j (buffer col j*16),
      // second WMMA reads tile j+1 (K=0..15 live, K=16..31 multiplied by zero B).
      const _Float16* ap = vbox + ln * VB_STRIDE + j * 16 + lh * 16;
      h8 a0lo = *(const h8*)(ap);
      h8 a0hi = *(const h8*)(ap + 8);
      h8 a1lo = *(const h8*)(ap + 32);
      h8 a1hi = *(const h8*)(ap + 40);
      h16 aLC = __builtin_shufflevector(a0lo, a0hi, 0,1,2,3,4,5,6,7,8,9,10,11,12,13,14,15);
      h16 aR  = __builtin_shufflevector(a1lo, a1hi, 0,1,2,3,4,5,6,7,8,9,10,11,12,13,14,15);
      f8 c = {0.f,0.f,0.f,0.f,0.f,0.f,0.f,0.f};
      c = __builtin_amdgcn_wmma_f32_16x16x32_f16(false, aLC, false, bH1, (short)0, c, false, false);
      c = __builtin_amdgcn_wmma_f32_16x16x32_f16(false, aR,  false, bH2, (short)0, c, false, false);
      hc[tt] = c;
    }
    const float dF = (float)d;
#pragma unroll
    for (int v = 0; v < 8; ++v) {  // strict < : first-min semantics like jnp.argmin
      if (hc[0][v] < bestc0[v]) { bestc0[v] = hc[0][v]; bestd0[v] = dF; }
      if (hc[1][v] < bestc1[v]) { bestc1[v] = hc[1][v]; bestd1[v] = dF; }
    }
    __syncthreads();
  }

  // ---- Emit per-group best (cost, disparity) for the cross-group merge ----
  float* cO = costOut + ((size_t)g * NPLANE + p) * PLANE_PIX;
  float* dO = dOut    + ((size_t)g * NPLANE + p) * PLANE_PIX;
#pragma unroll
  for (int v = 0; v < 8; ++v) {
    const int h = h0 + v + lh * 8;
    const int w0 = (wid * 2 + 0) * 16 + ln;
    const int w1 = (wid * 2 + 1) * 16 + ln;
    cO[h * WW + w0] = bestc0[v];
    dO[h * WW + w0] = bestd0[v];
    cO[h * WW + w1] = bestc1[v];
    dO[h * WW + w1] = bestd1[v];
  }
}

// ---------------------------------------------------------------------------
// Stage 3: merge the two disparity groups (strict < keeps lower d on ties).
// ---------------------------------------------------------------------------
__global__ __launch_bounds__(256)
void merge_kernel(const float* __restrict__ cost, const float* __restrict__ bd,
                  float* __restrict__ disp) {
  int i = blockIdx.x * blockDim.x + threadIdx.x;
  const int N = NPLANE * PLANE_PIX;
  if (i >= N) return;
  float c0 = cost[i], c1 = cost[N + i];
  disp[i] = (c1 < c0) ? bd[N + i] : bd[i];
}

// ---------------------------------------------------------------------------
// Stage 4: deterministic two-stage mean |disp_x - disp_gt| reduction.
// ---------------------------------------------------------------------------
__global__ __launch_bounds__(256)
void reduce_partial_kernel(const float* __restrict__ disp, float* __restrict__ partial) {
  __shared__ float sm[256];
  const int N = (NPLANE / 2) * PLANE_PIX;  // pixels per result tensor (x vs gt)
  float s = 0.f;
  for (int i = blockIdx.x * 256 + threadIdx.x; i < N; i += gridDim.x * 256)
    s += fabsf(disp[i] - disp[N + i]);
  sm[threadIdx.x] = s;
  __syncthreads();
  for (int off = 128; off > 0; off >>= 1) {
    if (threadIdx.x < off) sm[threadIdx.x] += sm[threadIdx.x + off];
    __syncthreads();
  }
  if (threadIdx.x == 0) partial[blockIdx.x] = sm[0];
}

__global__ __launch_bounds__(256)
void finalize_kernel(const float* __restrict__ partial, float* __restrict__ out) {
  __shared__ float sm[256];
  float s = 0.f;
  for (int i = threadIdx.x; i < NPART; i += 256) s += partial[i];
  sm[threadIdx.x] = s;
  __syncthreads();
  for (int off = 128; off > 0; off >>= 1) {
    if (threadIdx.x < off) sm[threadIdx.x] += sm[threadIdx.x + off];
    __syncthreads();
  }
  if (threadIdx.x == 0) out[0] = sm[0] / (float)((NPLANE / 2) * PLANE_PIX);
}

// ---------------------------------------------------------------------------
extern "C" void kernel_launch(void* const* d_in, const int* in_sizes, int n_in,
                              void* d_out, int out_size, void* d_ws, size_t ws_size,
                              hipStream_t stream) {
  (void)in_sizes; (void)n_in; (void)out_size; (void)ws_size;
  const float* x  = (const float*)d_in[0];
  const float* gt = (const float*)d_in[1];
  char* ws = (char*)d_ws;

  const size_t lpBytes   = (size_t)NPLANE * HH * WW * 4 * sizeof(_Float16);    // 5.2 MB
  const size_t rpBytes   = (size_t)NPLANE * HH * WPAD * 4 * sizeof(_Float16);  // 5.9 MB
  const size_t costBytes = (size_t)DGROUPS * NPLANE * PLANE_PIX * sizeof(float);
  const size_t dispBytes = (size_t)NPLANE * PLANE_PIX * sizeof(float);

  h4*    Lp   = (h4*)ws;
  h4*    Rp   = (h4*)(ws + lpBytes);
  float* cost = (float*)(ws + lpBytes + rpBytes);
  float* bd   = (float*)(ws + lpBytes + rpBytes + costBytes);
  float* disp = (float*)(ws + lpBytes + rpBytes + 2 * costBytes);
  float* part = (float*)(ws + lpBytes + rpBytes + 2 * costBytes + dispBytes);

  normalize_kernel<<<(IMGS * BATCH * HH * WW + 255) / 256, 256, 0, stream>>>(x, gt, Lp, Rp);
  bm_cost_kernel<<<DGROUPS * NPLANE * ROW_TILES, 512, SMEM_BYTES, stream>>>(Lp, Rp, cost, bd);
  merge_kernel<<<(NPLANE * PLANE_PIX + 255) / 256, 256, 0, stream>>>(cost, bd, disp);
  reduce_partial_kernel<<<NPART, 256, 0, stream>>>(disp, part);
  finalize_kernel<<<1, 256, 0, stream>>>(part, (float*)d_out);
}
